// DotProductAttentionHead_32693291057696
// MI455X (gfx1250) — compile-verified, matrix operand order checked by reading, not run on hardware
//
#include <hip/hip_runtime.h>
#include <math.h>

typedef __attribute__((ext_vector_type(16))) __bf16 v16bf;
typedef __attribute__((ext_vector_type(8)))  float  v8f;
typedef __attribute__((ext_vector_type(4)))  unsigned int u32x4;
typedef __attribute__((ext_vector_type(8)))  int          i32x8;
typedef __attribute__((ext_vector_type(4)))  int          i32x4;

#define BATCH 8
#define L1D   2048
#define L2D   2048
#define HD    512
#define BQ    64          // query rows per workgroup
#define QP    528         // bf16 pitch for Q/K rows in LDS (1056 B, %32==0)

// ---- TDM kernel tiling ----
#define TBK     32                      // keys per KV tile (double buffered)
#define TVP     48                      // bf16 pitch for Vt/P rows (96 B, %32==0)
#define KB_TILE (TBK * QP * 2)          // 33792 B per K buffer
#define VB_TILE (HD * TVP * 2)          // 49152 B per Vt buffer
#define OFF_K   (BQ * QP * 2)           // 67584
#define OFF_V   (OFF_K + 2 * KB_TILE)   // 135168
#define OFF_P   (OFF_V + 2 * VB_TILE)   // 233472
#define OFF_S   (OFF_P + BQ * TVP * 2)  // 239616
#define SMEM_TDM (OFF_S + 448 * 4)      // 241408 B  (< 320 KB WGP LDS)

// ---- fallback kernel tiling (round-2 proven path) ----
#define FBK 64
#define FVP 80

// ===================== TDM descriptor helper (ISA ch.8 D# layout) ============
// 2D tile load: tile_d0 contiguous elements per row, tile_d1 rows,
// row stride = stride0 elements, bf16 (data_size=1 -> 2B),
// LDS padding: pad_interval/pad_amount codes per ISA group1 tables.
__device__ __forceinline__ void tdm_load_2d(unsigned lds_off,
                                            unsigned long long ga,
                                            unsigned tile_d0, unsigned tile_d1,
                                            unsigned stride0,
                                            unsigned pad_int_code,
                                            unsigned pad_amt_code) {
  union { unsigned u[4]; u32x4 v; } g0;
  g0.u[0] = 1u;                                   // count=1, user descriptor
  g0.u[1] = lds_off;                              // lds_addr (bytes)
  g0.u[2] = (unsigned)ga;                         // global_addr[31:0]
  g0.u[3] = (unsigned)((ga >> 32) & 0x1FFFFFFu)   // global_addr[56:32]
          | (2u << 30);                           // type = 2 ("image")
  union { int i[8]; i32x8 v; } g1;
  g1.i[0] = (int)((1u << 16)                      // data_size = 2 bytes
                | (1u << 20)                      // pad_enable
                | (pad_int_code << 22)
                | (pad_amt_code << 25));
  // tensor_dim0 = tile_d0 (bits 79:48), tensor_dim1 = tile_d1 (bits 111:80)
  g1.i[1] = (int)((tile_d0 & 0xFFFFu) << 16);                   // barrier addr=0
  g1.i[2] = (int)((tile_d0 >> 16) | ((tile_d1 & 0xFFFFu) << 16));
  g1.i[3] = (int)((tile_d1 >> 16) | (tile_d0 << 16));           // tile_dim0
  g1.i[4] = (int)(tile_d1 & 0xFFFFu);                           // tile_dim1
  g1.i[5] = (int)stride0;                                       // dim0 stride lo
  g1.i[6] = 0;
  g1.i[7] = 0;
  i32x4 z4 = {0, 0, 0, 0};
  i32x8 z8 = {0, 0, 0, 0, 0, 0, 0, 0};
  // amdgpu-toolchain (clang-23) 6-arg form: (g0, g1, g2, g3, g4, cpol)
  __builtin_amdgcn_tensor_load_to_lds(g0.v, g1.v, z4, z4, z8, 0);
}

// ===================== pre-pass: fp32 -> bf16 (row major) ====================
__global__ __launch_bounds__(256)
void conv_bf16_kernel(const float* __restrict__ src, __bf16* __restrict__ dst) {
  const size_t i = ((size_t)blockIdx.x * 256 + threadIdx.x) * 8;
  float4 a = *(const float4*)(src + i);
  float4 b = *(const float4*)(src + i + 4);
  union { __bf16 h[8]; uint4 u; } t;
  t.h[0] = (__bf16)a.x; t.h[1] = (__bf16)a.y; t.h[2] = (__bf16)a.z; t.h[3] = (__bf16)a.w;
  t.h[4] = (__bf16)b.x; t.h[5] = (__bf16)b.y; t.h[6] = (__bf16)b.z; t.h[7] = (__bf16)b.w;
  *(uint4*)(dst + i) = t.u;
}

// ============ pre-pass: V[b][l][h] fp32 -> Vt[b][h][l] bf16 (tiled) ==========
__global__ __launch_bounds__(256)
void transpose_v_kernel(const float* __restrict__ V, __bf16* __restrict__ Vt) {
  __shared__ __bf16 T[64 * 72];
  const int tid = threadIdx.x;
  const int b   = blockIdx.y;
  const int l0  = (blockIdx.x & 31) * 64;   // L2D/64 = 32 tiles along l
  const int h0  = (blockIdx.x >> 5) * 64;   // HD/64  =  8 tiles along h
#pragma unroll
  for (int i = 0; i < 4; ++i) {             // 64x64 floats = 1024 float4
    int idx = tid + i * 256;
    int l   = idx >> 4;
    int h4  = (idx & 15) << 2;
    float4 v = *(const float4*)(V + ((size_t)(b * L2D + l0 + l)) * HD + h0 + h4);
    union { __bf16 h[4]; unsigned long long u; } t;
    t.h[0] = (__bf16)v.x; t.h[1] = (__bf16)v.y; t.h[2] = (__bf16)v.z; t.h[3] = (__bf16)v.w;
    *(unsigned long long*)(T + l * 72 + h4) = t.u;
  }
  __syncthreads();
#pragma unroll
  for (int i = 0; i < 4; ++i) {
    int idx = tid + i * 256;
    int h   = idx >> 4;
    int l4  = (idx & 15) << 2;
    union { __bf16 h[4]; unsigned long long u; } t;
    t.h[0] = T[(l4 + 0) * 72 + h];
    t.h[1] = T[(l4 + 1) * 72 + h];
    t.h[2] = T[(l4 + 2) * 72 + h];
    t.h[3] = T[(l4 + 3) * 72 + h];
    *(unsigned long long*)(Vt + ((size_t)(b * HD + h0 + h)) * L2D + l0 + l4) = t.u;
  }
}

// ===================== main attention kernel (TDM staged) ====================
__global__ __launch_bounds__(256)
void attn_fwd_tdm(const __bf16* __restrict__ Qb,
                  const __bf16* __restrict__ Kb,
                  const __bf16* __restrict__ Vt,
                  const long long* __restrict__ lens,
                  float* __restrict__ Og) {
  extern __shared__ char smem_raw[];
  char* smem = (char*)__builtin_assume_aligned(smem_raw, 32);

  __bf16* Qsm   = (__bf16*)smem;
  __bf16* Ksm0  = (__bf16*)(smem + OFF_K);
  __bf16* Vsm0  = (__bf16*)(smem + OFF_V);
  __bf16* Psm   = (__bf16*)(smem + OFF_P);
  float*  Mst   = (float*)(smem + OFF_S);
  float*  Lst   = Mst + BQ;
  float*  Ast   = Lst + BQ;
  float*  pmax  = Ast + BQ;    // 8 tiles * 16 rows
  float*  psum  = pmax + 128;  // 8 tiles * 16 rows

  const int tid  = threadIdx.x;
  const int w    = tid >> 5;
  const int lane = tid & 31;
  const int half = lane >> 4;
  const int lcol = lane & 15;
  const int mi   = w >> 1;     // score-phase query sub-block 0..3
  const int ni   = w & 1;      // score-phase key sub-block 0..1

  const int b     = blockIdx.y;
  const int qbase = blockIdx.x * BQ;
  const int mlen  = (int)lens[b];
  const int ntile = (mlen + TBK - 1) / TBK;

  const float scale = 0.044194173824159216f; // 1/sqrt(512)

  // ---- issue TDM loads: Q tile + first K/Vt tiles ----
  if (w == 0) {
    tdm_load_2d(0u, (unsigned long long)(Qb + ((size_t)(b * L1D + qbase)) * HD),
                HD, BQ, HD, 7u, 7u);                       // 1024B rows +32B pad
    tdm_load_2d(OFF_K, (unsigned long long)(Kb + ((size_t)(b * L2D)) * HD),
                HD, TBK, HD, 7u, 7u);
    tdm_load_2d(OFF_V, (unsigned long long)(Vt + (size_t)b * HD * L2D),
                TBK, HD, L2D, 3u, 7u);                     // 64B rows +32B pad
    __builtin_amdgcn_s_wait_tensorcnt(0);
  }
  if (tid < BQ) { Mst[tid] = -__builtin_inff(); Lst[tid] = 0.0f; }

  v8f Oacc[4][4];
#pragma unroll
  for (int m = 0; m < 4; ++m)
#pragma unroll
    for (int hs = 0; hs < 4; ++hs) Oacc[m][hs] = (v8f){};

  __syncthreads();

  for (int j = 0; j < ntile; ++j) {
    const int p = j & 1;
    const __bf16* Ksm = (const __bf16*)((const char*)Ksm0 + p * KB_TILE);
    const __bf16* Vsm = (const __bf16*)((const char*)Vsm0 + p * VB_TILE);

    // ---- prefetch next KV tile into the other buffer (async, no wait) ----
    if (w == 0 && j + 1 < ntile) {
      const int q = p ^ 1;
      tdm_load_2d(OFF_K + q * KB_TILE,
                  (unsigned long long)(Kb + ((size_t)(b * L2D + (j + 1) * TBK)) * HD),
                  HD, TBK, HD, 7u, 7u);
      tdm_load_2d(OFF_V + q * VB_TILE,
                  (unsigned long long)(Vt + (size_t)b * HD * L2D + (size_t)(j + 1) * TBK),
                  TBK, HD, L2D, 3u, 7u);
    }

    // ---- phase S: one 16x16 score tile per wave, S = (Q K^T) * scale ----
    v8f S = (v8f){};
    {
      const __bf16* qp = Qsm + (mi * 16 + lcol) * QP + half * 16;
      const __bf16* kp = Ksm + (ni * 16 + lcol) * QP + half * 16;
#pragma unroll
      for (int c = 0; c < 16; ++c) {
        v16bf a  = *(const v16bf*)(qp + c * 32);
        v16bf bb = *(const v16bf*)(kp + c * 32);
        S = __builtin_amdgcn_wmma_f32_16x16x32_bf16(false, a, false, bb,
                                                    (short)0, S, false, false);
      }
    }
    {
      const int kglob = j * TBK + ni * 16 + lcol;
      const bool dead = kglob >= mlen;
#pragma unroll
      for (int v = 0; v < 8; ++v)
        S[v] = dead ? -__builtin_inff() : S[v] * scale;
    }

    // ---- per-row partial max ----
    {
      float rmax[8];
#pragma unroll
      for (int v = 0; v < 8; ++v) rmax[v] = S[v];
#pragma unroll
      for (int off = 1; off < 16; off <<= 1)
#pragma unroll
        for (int v = 0; v < 8; ++v)
          rmax[v] = fmaxf(rmax[v], __shfl_xor(rmax[v], off, 32));
      if (lcol == 0)
#pragma unroll
        for (int v = 0; v < 8; ++v)
          pmax[(mi * 2 + ni) * 16 + half * 8 + v] = rmax[v];
    }
    __syncthreads();

    // ---- online-softmax stat update ----
    if (tid < BQ) {
      const int r  = tid;
      const int rl = r & 15;
      const int t0 = (r >> 4) * 2;
      float mj   = fmaxf(pmax[(t0 + 0) * 16 + rl], pmax[(t0 + 1) * 16 + rl]);
      float mold = Mst[r];
      float mnew = fmaxf(mold, mj);
      Mst[r] = mnew;
      Ast[r] = __expf(mold - mnew);
    }
    __syncthreads();

    // ---- phase P: exp(S - m) -> bf16 LDS, partial row sums ----
    {
      float pr[8];
#pragma unroll
      for (int v = 0; v < 8; ++v) {
        const int rl = mi * 16 + half * 8 + v;
        pr[v] = __expf(S[v] - Mst[rl]);
        Psm[rl * TVP + ni * 16 + lcol] = (__bf16)pr[v];
      }
#pragma unroll
      for (int off = 1; off < 16; off <<= 1)
#pragma unroll
        for (int v = 0; v < 8; ++v)
          pr[v] += __shfl_xor(pr[v], off, 32);
      if (lcol == 0)
#pragma unroll
        for (int v = 0; v < 8; ++v)
          psum[(mi * 2 + ni) * 16 + half * 8 + v] = pr[v];
    }
    __syncthreads();

    if (tid < BQ) {
      const int r  = tid;
      const int rl = r & 15;
      const int t0 = (r >> 4) * 2;
      Lst[r] = Lst[r] * Ast[r] + psum[(t0 + 0) * 16 + rl] + psum[(t0 + 1) * 16 + rl];
    }

    // ---- phase O: rescale + O += P @ V (wave owns 64 H-cols) ----
#pragma unroll
    for (int m = 0; m < 4; ++m) {
      float av[8];
#pragma unroll
      for (int v = 0; v < 8; ++v) av[v] = Ast[m * 16 + half * 8 + v];
      v16bf a = *(const v16bf*)(Psm + (m * 16 + lcol) * TVP + half * 16);
#pragma unroll
      for (int hs = 0; hs < 4; ++hs) {
        v8f acc = Oacc[m][hs];
#pragma unroll
        for (int v = 0; v < 8; ++v) acc[v] *= av[v];
        v16bf bb = *(const v16bf*)(Vsm + (w * 64 + hs * 16 + lcol) * TVP + half * 16);
        acc = __builtin_amdgcn_wmma_f32_16x16x32_bf16(false, a, false, bb,
                                                      (short)0, acc, false, false);
        Oacc[m][hs] = acc;
      }
    }

    if (w == 0) __builtin_amdgcn_s_wait_tensorcnt(0);  // next tiles landed
    __syncthreads();
  }

  // ---- epilogue ----
#pragma unroll
  for (int m = 0; m < 4; ++m) {
    float linv[8];
#pragma unroll
    for (int v = 0; v < 8; ++v)
      linv[v] = 1.0f / Lst[m * 16 + half * 8 + v];
#pragma unroll
    for (int hs = 0; hs < 4; ++hs)
#pragma unroll
      for (int v = 0; v < 8; ++v) {
        const int row = qbase + m * 16 + half * 8 + v;
        const int col = w * 64 + hs * 16 + lcol;
        Og[((size_t)(b * L1D + row)) * HD + col] = Oacc[m][hs][v] * linv[v];
      }
  }
}

// ===================== fallback kernel (round-2 path, no d_ws needed) ========
union BF4 { __bf16 h[4]; unsigned long long u64; };

__global__ __launch_bounds__(256)
void attn_fwd_fallback(const float* __restrict__ Qg,
                       const float* __restrict__ Kg,
                       const float* __restrict__ Vg,
                       const long long* __restrict__ lens,
                       float* __restrict__ Og) {
  extern __shared__ char smem_raw[];
  char* smem = (char*)__builtin_assume_aligned(smem_raw, 32);
  __bf16* Qsm = (__bf16*)smem;
  __bf16* Ksm = Qsm + BQ * QP;
  __bf16* Vsm = Ksm + FBK * QP;
  __bf16* Psm = Vsm + HD * FVP;
  float*  Mst = (float*)(Psm + BQ * FVP);
  float*  Lst = Mst + BQ;
  float*  Ast = Lst + BQ;
  float*  pmax = Ast + BQ;
  float*  psum = pmax + 256;

  const int tid = threadIdx.x, w = tid >> 5, lane = tid & 31;
  const int half = lane >> 4, lcol = lane & 15;
  const int miB = (w >> 2) * 2, ni = w & 3;
  const int b = blockIdx.y, qbase = blockIdx.x * BQ;
  const int mlen = (int)lens[b];
  const int ntile = (mlen + FBK - 1) / FBK;
  const float scale = 0.044194173824159216f;

  {
    const float* qsrc = Qg + ((size_t)(b * L1D + qbase)) * HD;
#pragma unroll
    for (int i = 0; i < 32; ++i) {
      int idx = tid + i * 256, row = idx >> 7, c4 = (idx & 127) << 2;
      float4 qv = *(const float4*)(qsrc + (size_t)row * HD + c4);
      BF4 t;
      t.h[0] = (__bf16)qv.x; t.h[1] = (__bf16)qv.y;
      t.h[2] = (__bf16)qv.z; t.h[3] = (__bf16)qv.w;
      *(unsigned long long*)(Qsm + row * QP + c4) = t.u64;
    }
    if (tid < BQ) { Mst[tid] = -__builtin_inff(); Lst[tid] = 0.0f; }
  }
  v8f Oacc[4][4];
#pragma unroll
  for (int m = 0; m < 4; ++m)
#pragma unroll
    for (int hs = 0; hs < 4; ++hs) Oacc[m][hs] = (v8f){};
  __syncthreads();

  for (int j = 0; j < ntile; ++j) {
    {
      const float* ksrc = Kg + ((size_t)(b * L2D + j * FBK)) * HD;
      const float* vsrc = Vg + ((size_t)(b * L2D + j * FBK)) * HD;
#pragma unroll
      for (int i = 0; i < 32; ++i) {
        int idx = tid + i * 256, row = idx >> 7, c4 = (idx & 127) << 2;
        float4 kv = *(const float4*)(ksrc + (size_t)row * HD + c4);
        BF4 t;
        t.h[0] = (__bf16)kv.x; t.h[1] = (__bf16)kv.y;
        t.h[2] = (__bf16)kv.z; t.h[3] = (__bf16)kv.w;
        *(unsigned long long*)(Ksm + row * QP + c4) = t.u64;
        float4 vv = *(const float4*)(vsrc + (size_t)row * HD + c4);
        Vsm[(c4 + 0) * FVP + row] = (__bf16)vv.x;
        Vsm[(c4 + 1) * FVP + row] = (__bf16)vv.y;
        Vsm[(c4 + 2) * FVP + row] = (__bf16)vv.z;
        Vsm[(c4 + 3) * FVP + row] = (__bf16)vv.w;
      }
      if (j + 1 < ntile) {
        const char* nk = (const char*)(Kg + ((size_t)(b * L2D + (j + 1) * FBK)) * HD);
        const char* nv = (const char*)(Vg + ((size_t)(b * L2D + (j + 1) * FBK)) * HD);
        __builtin_prefetch(nk + tid * 512, 0, 1);
        __builtin_prefetch(nv + tid * 512, 0, 1);
      }
    }
    __syncthreads();

    v8f S[2];
#pragma unroll
    for (int t = 0; t < 2; ++t) {
      const int mi = miB + t;
      v8f acc = (v8f){};
      const __bf16* qp = Qsm + (mi * 16 + lcol) * QP + half * 16;
      const __bf16* kp = Ksm + (ni * 16 + lcol) * QP + half * 16;
#pragma unroll
      for (int c = 0; c < 16; ++c) {
        v16bf a  = *(const v16bf*)(qp + c * 32);
        v16bf bb = *(const v16bf*)(kp + c * 32);
        acc = __builtin_amdgcn_wmma_f32_16x16x32_bf16(false, a, false, bb,
                                                      (short)0, acc, false, false);
      }
      S[t] = acc;
    }
    {
      const int kglob = j * FBK + ni * 16 + lcol;
      const bool dead = kglob >= mlen;
#pragma unroll
      for (int t = 0; t < 2; ++t)
#pragma unroll
        for (int v = 0; v < 8; ++v)
          S[t][v] = dead ? -__builtin_inff() : S[t][v] * scale;
    }
#pragma unroll
    for (int t = 0; t < 2; ++t) {
      const int mi = miB + t;
      float rmax[8];
#pragma unroll
      for (int v = 0; v < 8; ++v) rmax[v] = S[t][v];
#pragma unroll
      for (int off = 1; off < 16; off <<= 1)
#pragma unroll
        for (int v = 0; v < 8; ++v)
          rmax[v] = fmaxf(rmax[v], __shfl_xor(rmax[v], off, 32));
      if (lcol == 0)
#pragma unroll
        for (int v = 0; v < 8; ++v)
          pmax[(mi * 4 + ni) * 16 + half * 8 + v] = rmax[v];
    }
    __syncthreads();
    if (tid < BQ) {
      const int r = tid, rl = r & 15, t0 = (r >> 4) * 4;
      float mj = fmaxf(fmaxf(pmax[(t0 + 0) * 16 + rl], pmax[(t0 + 1) * 16 + rl]),
                       fmaxf(pmax[(t0 + 2) * 16 + rl], pmax[(t0 + 3) * 16 + rl]));
      float mold = Mst[r], mnew = fmaxf(mold, mj);
      Mst[r] = mnew;
      Ast[r] = __expf(mold - mnew);
    }
    __syncthreads();
#pragma unroll
    for (int t = 0; t < 2; ++t) {
      const int mi = miB + t;
      float p[8];
#pragma unroll
      for (int v = 0; v < 8; ++v) {
        const int rl = mi * 16 + half * 8 + v;
        p[v] = __expf(S[t][v] - Mst[rl]);
        Psm[rl * FVP + ni * 16 + lcol] = (__bf16)p[v];
      }
#pragma unroll
      for (int off = 1; off < 16; off <<= 1)
#pragma unroll
        for (int v = 0; v < 8; ++v)
          p[v] += __shfl_xor(p[v], off, 32);
      if (lcol == 0)
#pragma unroll
        for (int v = 0; v < 8; ++v)
          psum[(mi * 4 + ni) * 16 + half * 8 + v] = p[v];
    }
    __syncthreads();
    if (tid < BQ) {
      const int r = tid, rl = r & 15, t0 = (r >> 4) * 4;
      float s = psum[(t0 + 0) * 16 + rl] + psum[(t0 + 1) * 16 + rl] +
                psum[(t0 + 2) * 16 + rl] + psum[(t0 + 3) * 16 + rl];
      Lst[r] = Lst[r] * Ast[r] + s;
    }
#pragma unroll
    for (int m = 0; m < 4; ++m) {
      float av[8];
#pragma unroll
      for (int v = 0; v < 8; ++v) av[v] = Ast[m * 16 + half * 8 + v];
      const __bf16* pp = Psm + (m * 16 + lcol) * FVP + half * 16;
      v16bf a0 = *(const v16bf*)(pp);
      v16bf a1 = *(const v16bf*)(pp + 32);
#pragma unroll
      for (int hs = 0; hs < 4; ++hs) {
        v8f acc = Oacc[m][hs];
#pragma unroll
        for (int v = 0; v < 8; ++v) acc[v] *= av[v];
        const __bf16* vb = Vsm + (w * 64 + hs * 16 + lcol) * FVP + half * 16;
        v16bf b0 = *(const v16bf*)(vb);
        v16bf b1 = *(const v16bf*)(vb + 32);
        acc = __builtin_amdgcn_wmma_f32_16x16x32_bf16(false, a0, false, b0,
                                                      (short)0, acc, false, false);
        acc = __builtin_amdgcn_wmma_f32_16x16x32_bf16(false, a1, false, b1,
                                                      (short)0, acc, false, false);
        Oacc[m][hs] = acc;
      }
    }
    __syncthreads();
  }
#pragma unroll
  for (int m = 0; m < 4; ++m) {
    float linv[8];
#pragma unroll
    for (int v = 0; v < 8; ++v)
      linv[v] = 1.0f / Lst[m * 16 + half * 8 + v];
#pragma unroll
    for (int hs = 0; hs < 4; ++hs)
#pragma unroll
      for (int v = 0; v < 8; ++v) {
        const int row = qbase + m * 16 + half * 8 + v;
        const int col = w * 64 + hs * 16 + lcol;
        Og[((size_t)(b * L1D + row)) * HD + col] = Oacc[m][hs][v] * linv[v];
      }
  }
}

// ============================== launcher =====================================
extern "C" void kernel_launch(void* const* d_in, const int* in_sizes, int n_in,
                              void* d_out, int out_size, void* d_ws, size_t ws_size,
                              hipStream_t stream) {
  (void)in_sizes; (void)n_in; (void)out_size;
  const float*     Qg  = (const float*)d_in[0];
  const float*     Kg  = (const float*)d_in[1];
  const float*     Vg  = (const float*)d_in[2];
  const long long* len = (const long long*)d_in[3];
  float*           Og  = (float*)d_out;

  constexpr size_t N_ELEMS   = (size_t)BATCH * L1D * HD;     // 8,388,608 per tensor
  constexpr size_t BF_BYTES  = N_ELEMS * 2;                  // 16 MB per bf16 tensor
  constexpr size_t WS_NEEDED = 3 * BF_BYTES;                 // Qb + Kb + Vt

  if (ws_size >= WS_NEEDED) {
    __bf16* Qb  = (__bf16*)d_ws;
    __bf16* Kb  = Qb + N_ELEMS;
    __bf16* Vtb = Kb + N_ELEMS;
    // pre-pass: convert + transpose (one-time ~200 MB of HBM traffic)
    conv_bf16_kernel<<<dim3(N_ELEMS / (256 * 8)), dim3(256), 0, stream>>>(Qg, Qb);
    conv_bf16_kernel<<<dim3(N_ELEMS / (256 * 8)), dim3(256), 0, stream>>>(Kg, Kb);
    transpose_v_kernel<<<dim3((L2D / 64) * (HD / 64), BATCH), dim3(256), 0, stream>>>(Vg, Vtb);
    attn_fwd_tdm<<<dim3(L1D / BQ, BATCH), dim3(256), SMEM_TDM, stream>>>(Qb, Kb, Vtb, len, Og);
  } else {
    constexpr size_t smem =
        (size_t)(BQ * QP + FBK * QP + HD * FVP + BQ * FVP) * sizeof(__bf16) +
        (size_t)(3 * BQ + 512) * sizeof(float);
    attn_fwd_fallback<<<dim3(L1D / BQ, BATCH), dim3(256), smem, stream>>>(Qg, Kg, Vg, len, Og);
  }
}
